// DecoderGRU_31490700214516
// MI455X (gfx1250) — matrix operand order repaired and checked by reading, not compiled, not run on hardware
//
#include <hip/hip_runtime.h>
#include <hip/hip_bf16.h>
#include <math.h>

#define BSZ 4096
#define TSTEPS 64
#define HDIM 64
#define PDIM 50
#define VDIM 32
#define ROWS_WG 32
#define THREADS_MAIN 64

typedef __attribute__((ext_vector_type(16))) __bf16       v16bf;
typedef __attribute__((ext_vector_type(8)))  float        v8f;
typedef __attribute__((ext_vector_type(4)))  unsigned int v4u;

union V16 { v4u u[2]; v16bf v; __bf16 h[16]; };

__device__ __forceinline__ v8f wmma_bf16(v16bf a, v16bf b, v8f c) {
  // D = A(16x32 bf16) x B(32x16 bf16) + C(16x16 f32), wave32
  return __builtin_amdgcn_wmma_f32_16x16x32_bf16(false, a, false, b, (short)0, c, false, false);
}

// Native-rate activations: v_exp_f32 / v_rcp_f32 (TRANS ops, compiler-managed
// hazards, co-executable with WMMA). Avoids the IEEE v_div_* expansion.
__device__ __forceinline__ float sigmoidf_(float x) {
  return __builtin_amdgcn_rcpf(1.0f + __expf(-x));
}
__device__ __forceinline__ float tanh_fast(float x) {
  return fmaf(2.0f, __builtin_amdgcn_rcpf(1.0f + __expf(-2.0f * x)), -1.0f);
}

// One wave owns 16 batch rows for the entire T=64 recurrence.
// Block = 64 threads = 2 waves = 32 rows; grid = 128 blocks covers B=4096.
__global__ void __launch_bounds__(THREADS_MAIN, 1)
gru_fused_kernel(const float* __restrict__ phon, const int* __restrict__ cs,
                 const float* __restrict__ emb,
                 const float* __restrict__ Wrx, const float* __restrict__ brx,
                 const float* __restrict__ Wrh, const float* __restrict__ brh,
                 const float* __restrict__ Wzx, const float* __restrict__ bzx,
                 const float* __restrict__ Wzh, const float* __restrict__ bzh,
                 const float* __restrict__ Whx, const float* __restrict__ bhx,
                 const float* __restrict__ Whh, const float* __restrict__ bhh,
                 const float* __restrict__ Wproj, const float* __restrict__ bproj,
                 float* __restrict__ out) {
  __shared__ float  tableS[32 * 192];        // emb@W_*x + b_*x + b_*h  (vocab lookup)
  __shared__ float  phonS[ROWS_WG * 192];    // phon@W_*x[64:] (staging; hoisted to VGPRs)
  __shared__ int    charS[64 * ROWS_WG];     // [t][row], pre-scaled by 192
  __shared__ __bf16 stageS[2][16 * 64];      // per-wave h/rh transpose staging

  const int tid   = threadIdx.x;
  const int bBase = blockIdx.x * ROWS_WG;
  const int lane  = tid & 31;
  const int wave  = tid >> 5;
  const int hl    = lane >> 4;    // lane half (0/1)
  const int l15   = lane & 15;

  // ---- cooperative LDS init ----
  for (int i = tid; i < 64 * ROWS_WG; i += THREADS_MAIN) {
    int t = i >> 5, row = i & 31;
    charS[i] = cs[(bBase + row) * 65 + t] * 192;   // pre-scaled table row offset
  }

  const float* Wx[3]  = {Wrx, Wzx, Whx};
  const float* bxv[3] = {brx, bzx, bhx};
  const float* bhv[3] = {brh, bzh, bhh};
  for (int i = tid; i < 32 * 192; i += THREADS_MAIN) {
    int v = i / 192, gc = i % 192, g = gc >> 6, c = gc & 63;
    const float* W = Wx[g];
    float s = bxv[g][c] + bhv[g][c];
    for (int k = 0; k < 64; ++k) s += emb[v * 64 + k] * W[k * 64 + c];
    tableS[i] = s;
  }
  for (int i = tid; i < ROWS_WG * 192; i += THREADS_MAIN) {
    int row = i / 192, gc = i % 192, g = gc >> 6, c = gc & 63;
    const float* W = Wx[g];
    float s = 0.0f;
    for (int k = 0; k < PDIM; ++k) s += phon[(bBase + row) * PDIM + k] * W[(64 + k) * 64 + c];
    phonS[i] = s;
  }

  // ---- recurrent weights: bf16, pre-swizzled into WMMA B-operand layout, held in VGPRs ----
  // B(32x16): lane l -> column N=l%16, K = ks*32 + (j<8 ? (l>=16?8:0) + j : 16 + (l>=16?8:0) + j-8)
  v16bf Wr[4][2], Wz[4][2], Wh[4][2], Wp[2][2];
#pragma unroll
  for (int nt = 0; nt < 4; ++nt) {
#pragma unroll
    for (int ks = 0; ks < 2; ++ks) {
      V16 wr, wz, wh;
#pragma unroll
      for (int j = 0; j < 16; ++j) {
        int K = ks * 32 + (j < 8 ? hl * 8 + j : 16 + hl * 8 + (j - 8)), N = nt * 16 + l15;
        wr.h[j] = (__bf16)Wrh[K * 64 + N];
        wz.h[j] = (__bf16)Wzh[K * 64 + N];
        wh.h[j] = (__bf16)Whh[K * 64 + N];
      }
      Wr[nt][ks] = wr.v; Wz[nt][ks] = wz.v; Wh[nt][ks] = wh.v;
    }
  }
#pragma unroll
  for (int nt = 0; nt < 2; ++nt) {
#pragma unroll
    for (int ks = 0; ks < 2; ++ks) {
      V16 wp;
#pragma unroll
      for (int j = 0; j < 16; ++j) {
        int K = ks * 32 + (j < 8 ? hl * 8 + j : 16 + hl * 8 + (j - 8)), N = nt * 16 + l15;
        wp.h[j] = (__bf16)Wproj[K * 32 + N];
      }
      Wp[nt][ks] = wp.v;
    }
  }
  const float bp0 = bproj[l15], bp1 = bproj[16 + l15];
  __syncthreads();

  __bf16* stg = &stageS[wave][0];
  const int rowL = wave * 16 + hl * 8;   // this lane's first C/D row (local)
  const int hsb  = hl * 8;               // row base inside 16-row staging tile

  // ---- hoist t-invariant phonetic gate inputs into VGPRs (96 floats/lane) ----
  float phR[4][8], phZ[4][8], phC[4][8];
#pragma unroll
  for (int nt = 0; nt < 4; ++nt) {
#pragma unroll
    for (int r = 0; r < 8; ++r) {
      int p = (rowL + r) * 192 + nt * 16 + l15;
      phR[nt][r] = phonS[p];
      phZ[nt][r] = phonS[p + 64];
      phC[nt][r] = phonS[p + 128];
    }
  }

  v8f h[4];
#pragma unroll
  for (int nt = 0; nt < 4; ++nt) h[nt] = (v8f){0, 0, 0, 0, 0, 0, 0, 0};
  V16 a0, a1;                            // A-layout of h (bf16); h0 == 0
  a0.u[0] = a0.u[1] = (v4u){0, 0, 0, 0};
  a1.u[0] = a1.u[1] = (v4u){0, 0, 0, 0};

#pragma unroll 1
  for (int t = 0; t < TSTEPS; ++t) {
    // pre-scaled char table offsets for this lane's 8 rows: 2 x ds_load_b128
    v4u cA = *(const v4u*)&charS[t * 32 + rowL];
    v4u cB = *(const v4u*)&charS[t * 32 + rowL + 4];
    int cidx[8];
    cidx[0] = (int)cA.x; cidx[1] = (int)cA.y; cidx[2] = (int)cA.z; cidx[3] = (int)cA.w;
    cidx[4] = (int)cB.x; cidx[5] = (int)cB.y; cidx[6] = (int)cB.z; cidx[7] = (int)cB.w;

    // ---- r + z gates (paired table fetches -> ds_load_2addr) ----
    v8f zac[4];
#pragma unroll
    for (int nt = 0; nt < 4; ++nt) {
      int col = nt * 16 + l15;
      v8f accr, accz;
#pragma unroll
      for (int r = 0; r < 8; ++r) {
        int base = cidx[r] + col;
        accr[r] = tableS[base] + phR[nt][r];
        accz[r] = tableS[base + 64] + phZ[nt][r];
      }
      accr = wmma_bf16(a0.v, Wr[nt][0], accr);
      accr = wmma_bf16(a1.v, Wr[nt][1], accr);
      accz = wmma_bf16(a0.v, Wz[nt][0], accz);
      accz = wmma_bf16(a1.v, Wz[nt][1], accz);
#pragma unroll
      for (int r = 0; r < 8; ++r) {
        float rv = sigmoidf_(accr[r]);
        stg[(hsb + r) * 64 + col] = (__bf16)(rv * h[nt][r]);   // stage r*h
        zac[nt][r] = sigmoidf_(accz[r]);
      }
    }

    // ---- load A-layout of (r*h): per-wave staging, DS in-order per wave ----
    asm volatile("s_wait_dscnt 0" ::: "memory");
    V16 ar0, ar1;
    ar0.u[0] = *(const v4u*)&stg[l15 * 64 + hl * 8];
    ar0.u[1] = *(const v4u*)&stg[l15 * 64 + 16 + hl * 8];
    ar1.u[0] = *(const v4u*)&stg[l15 * 64 + 32 + hl * 8];
    ar1.u[1] = *(const v4u*)&stg[l15 * 64 + 48 + hl * 8];

    // ---- candidate + state update, restage h_new ----
#pragma unroll
    for (int nt = 0; nt < 4; ++nt) {
      int col = nt * 16 + l15;
      v8f acc;
#pragma unroll
      for (int r = 0; r < 8; ++r)
        acc[r] = tableS[cidx[r] + 128 + col] + phC[nt][r];
      acc = wmma_bf16(ar0.v, Wh[nt][0], acc);
      acc = wmma_bf16(ar1.v, Wh[nt][1], acc);
#pragma unroll
      for (int r = 0; r < 8; ++r) {
        float cc = tanh_fast(acc[r]);
        float z  = zac[nt][r];
        float hn = fmaf(z, cc - h[nt][r], h[nt][r]);   // (1-z)h + zc
        h[nt][r] = hn;
        stg[(hsb + r) * 64 + col] = (__bf16)hn;
      }
    }
    asm volatile("s_wait_dscnt 0" ::: "memory");
    a0.u[0] = *(const v4u*)&stg[l15 * 64 + hl * 8];
    a0.u[1] = *(const v4u*)&stg[l15 * 64 + 16 + hl * 8];
    a1.u[0] = *(const v4u*)&stg[l15 * 64 + 32 + hl * 8];
    a1.u[1] = *(const v4u*)&stg[l15 * 64 + 48 + hl * 8];

    // ---- logits = h_new @ W_proj + b_proj ----
#pragma unroll
    for (int nt = 0; nt < 2; ++nt) {
      float bpv = nt ? bp1 : bp0;
      v8f acc = {bpv, bpv, bpv, bpv, bpv, bpv, bpv, bpv};
      acc = wmma_bf16(a0.v, Wp[nt][0], acc);
      acc = wmma_bf16(a1.v, Wp[nt][1], acc);
#pragma unroll
      for (int r = 0; r < 8; ++r) {
        int b = bBase + rowL + r;
        out[((size_t)b * 64 + t) * 32 + nt * 16 + l15] = acc[r];
      }
    }
  }
}

// Deterministic masked-NLL reduction: stage 1 -> 256 partial (sum, count) pairs.
__global__ void loss_partial_kernel(const float* __restrict__ logits,
                                    const int* __restrict__ cs,
                                    float* __restrict__ partials) {
  __shared__ float ssum[256], scnt[256];
  const int tid = threadIdx.x;
  float s = 0.0f, c = 0.0f;
  for (int i = blockIdx.x * 256 + tid; i < BSZ * TSTEPS; i += 256 * 256) {
    int b = i >> 6, t = i & 63;
    int tgt = cs[b * 65 + t + 1];
    const float* row = logits + (size_t)i * 32;
    float m = row[0];
    for (int v = 1; v < 32; ++v) m = fmaxf(m, row[v]);
    float se = 0.0f;
    for (int v = 0; v < 32; ++v) se += __expf(row[v] - m);
    float lse = m + __logf(se);
    if (tgt != 0) { s += lse - row[tgt]; c += 1.0f; }
  }
  ssum[tid] = s; scnt[tid] = c;
  __syncthreads();
  for (int off = 128; off > 0; off >>= 1) {
    if (tid < off) { ssum[tid] += ssum[tid + off]; scnt[tid] += scnt[tid + off]; }
    __syncthreads();
  }
  if (tid == 0) { partials[blockIdx.x] = ssum[0]; partials[256 + blockIdx.x] = scnt[0]; }
}

__global__ void loss_final_kernel(const float* __restrict__ partials,
                                  float* __restrict__ out_loss) {
  __shared__ float ssum[256], scnt[256];
  const int tid = threadIdx.x;
  ssum[tid] = partials[tid];
  scnt[tid] = partials[256 + tid];
  __syncthreads();
  for (int off = 128; off > 0; off >>= 1) {
    if (tid < off) { ssum[tid] += ssum[tid + off]; scnt[tid] += scnt[tid + off]; }
    __syncthreads();
  }
  if (tid == 0) out_loss[0] = ssum[0] / fmaxf(scnt[0], 1.0f);
}

extern "C" void kernel_launch(void* const* d_in, const int* in_sizes, int n_in,
                              void* d_out, int out_size, void* d_ws, size_t ws_size,
                              hipStream_t stream) {
  const float* phon  = (const float*)d_in[0];
  const int*   cs    = (const int*)d_in[1];
  const float* emb   = (const float*)d_in[2];
  const float* Wrx   = (const float*)d_in[3];
  const float* brx   = (const float*)d_in[4];
  const float* Wrh   = (const float*)d_in[5];
  const float* brh   = (const float*)d_in[6];
  const float* Wzx   = (const float*)d_in[7];
  const float* bzx   = (const float*)d_in[8];
  const float* Wzh   = (const float*)d_in[9];
  const float* bzh   = (const float*)d_in[10];
  const float* Whx   = (const float*)d_in[11];
  const float* bhx   = (const float*)d_in[12];
  const float* Whh   = (const float*)d_in[13];
  const float* bhh   = (const float*)d_in[14];
  const float* Wproj = (const float*)d_in[15];
  const float* bproj = (const float*)d_in[16];
  float* out = (float*)d_out;
  float* partials = (float*)d_ws;  // 512 floats

  gru_fused_kernel<<<BSZ / ROWS_WG, THREADS_MAIN, 0, stream>>>(
      phon, cs, emb, Wrx, brx, Wrh, brh, Wzx, bzx, Wzh, bzh,
      Whx, bhx, Whh, bhh, Wproj, bproj, out);
  loss_partial_kernel<<<256, 256, 0, stream>>>(out, cs, partials);
  loss_final_kernel<<<1, 256, 0, stream>>>(partials, out + (size_t)BSZ * TSTEPS * VDIM);
}
